// MultiHeadAttentionQuantum_65481071409682
// MI455X (gfx1250) — matrix operand order repaired and checked by reading, not compiled, not run on hardware
//
#include <hip/hip_runtime.h>
#include <hip/hip_bf16.h>

// ---------------------------------------------------------------------------
// MultiHeadAttentionQuantum for MI455X (gfx1250), wave32 + WMMA bf16 + TDM.
//
// Math:  proj = x @ Wp^T + bp ; angles = proj + theta (per 8-group)
//        CNOT ring is GF(2)-linear  =>  expz[w] = prefix/suffix products of
//        cos(angle)  (no 256-dim state needed)
//        out  = expz @ Wc^T + bc
//
// Precision: bf16 WMMA (v_wmma_f32_16x16x32_bf16) with f32 accumulation.
// Data movement: W tiles staged to LDS by the Tensor Data Mover
// (tensor_load_to_lds, TENSORcnt), double buffered so the DMA overlaps WMMA.
// ---------------------------------------------------------------------------

typedef __attribute__((ext_vector_type(16))) __bf16 v16bf;
typedef __attribute__((ext_vector_type(8)))  float  v8f;
typedef __attribute__((ext_vector_type(4)))  unsigned int uint32x4;
typedef __attribute__((ext_vector_type(8)))  int          int32x8;
typedef __attribute__((ext_vector_type(4)))  int          int32x4;

#if defined(__has_builtin)
#  if __has_builtin(__builtin_amdgcn_tensor_load_to_lds)
#    define USE_TDM 1
#  endif
#endif
#ifndef USE_TDM
#  define USE_TDM 0
#endif

union BF16x16 {
    v16bf        v;
    unsigned int u[8];
};

__device__ __forceinline__ unsigned int pack2bf(float a, float b) {
    union { __bf16 h[2]; unsigned int u; } p;
    p.h[0] = (__bf16)a;
    p.h[1] = (__bf16)b;
    return p.u;
}

#define MDIM 4096   // B*S tokens
#define KDIM 768
#define NDIM 768
#define MT   128    // block M tile (8 waves x 16 rows)
#define NT   64     // block N tile (4 wmma tiles per wave)
#define KT   32     // K step (one bf16 WMMA K)

#if USE_TDM
// Issue a TDM 2D-tile load: gsrc -> LDS (NT rows x KT f32, row stride KDIM f32).
// D# fields per CDNA5 ISA §8.3/§8.4 (group0: count/lds/global/type,
// group1: data_size, tensor dims, tile dims, dim0 stride).
__device__ __forceinline__ void tdm_load_w_tile(const float* gsrc, void* ldst)
{
    unsigned long long ga = (unsigned long long)(size_t)gsrc;
    unsigned int       la = (unsigned int)(size_t)ldst;   // LDS offset = addr[31:0]

    uint32x4 g0 = {
        1u,                                              // count=1 (valid), user desc
        la,                                              // lds_addr (bytes)
        (unsigned int)(ga & 0xffffffffu),                // global_addr[31:0]
        (unsigned int)((ga >> 32) & 0x01ffffffu) | (2u << 30) // addr[56:32] | type=2
    };
    int32x8 g1 = {
        (int)(2u << 16),           // workgroup_mask=0, data_size=2 (4B), no flags
        (int)((unsigned)KDIM << 16), // atomic_barrier_addr=0 | tensor_dim0[15:0]=768
        (int)((unsigned)KDIM << 16), // tensor_dim0[31:16]=0 | tensor_dim1[15:0]=768
        (int)((unsigned)KT << 16),   // tensor_dim1[31:16]=0 | tile_dim0=32
        (int)NT,                     // tile_dim1=64 | tile_dim2=0
        (int)KDIM,                   // tensor_dim0_stride[31:0]=768
        0,                           // dim0_stride[47:32]=0 | dim1_stride[15:0]=0
        0                            // tensor_dim1_stride[47:16]=0 (2D tile)
    };
    int32x4 z4 = { 0, 0, 0, 0 };
#if __clang_major__ >= 23
    int32x8 z8 = { 0, 0, 0, 0, 0, 0, 0, 0 };
    __builtin_amdgcn_tensor_load_to_lds(g0, g1, z4, z4, z8, 0);
#else
    __builtin_amdgcn_tensor_load_to_lds(g0, g1, z4, z4, 0);
#endif
}
#endif // USE_TDM

// out[m,n] = sum_k A[m,k] * W[n,k] + bias[n]      (torch Linear: x @ W^T + b)
__global__ __launch_bounds__(256)
void gemm_bias_wmma(const float* __restrict__ A,
                    const float* __restrict__ W,
                    const float* __restrict__ bias,
                    float* __restrict__ out)
{
#if USE_TDM
    __shared__ float ldsWf[2][NT * KT];           // 2 x 8 KB f32 TDM landing pads
#endif
    __shared__ unsigned int ldsWb[NT * (KT / 2)]; // 4 KB packed bf16 (64 x 16 uints)

    const int tid  = threadIdx.x;
    const int wave = tid >> 5;       // 0..7
    const int lane = tid & 31;
    const int lo16 = lane & 15;
    const int hi   = lane >> 4;      // 0 or 1 (half-wave)

    const int m0 = blockIdx.y * MT + wave * 16;  // this wave's 16 output rows
    const int n0 = blockIdx.x * NT;

    v8f acc[4];
#pragma unroll
    for (int t = 0; t < 4; ++t)
#pragma unroll
        for (int i = 0; i < 8; ++i) acc[t][i] = 0.0f;

    // A-matrix 16x32 bf16 layout (ISA 7.12.2): lanes 0-15 & 16-31 both hold
    // row M = lane%16; per lane K chunks [hi*8, hi*8+8) and [16+hi*8, +8).
    const float* arow = A + (size_t)(m0 + lo16) * KDIM;

#if USE_TDM
    if (wave == 0)  // one DMA per workgroup: prime buffer 0 with the k=0 tile
        tdm_load_w_tile(W + (size_t)n0 * KDIM, &ldsWf[0][0]);
#endif

    int buf = 0;
    for (int k = 0; k < KDIM; k += KT, buf ^= 1) {
#if USE_TDM
        if (wave == 0)
            __builtin_amdgcn_s_wait_tensorcnt(0);   // current tile landed
        __syncthreads();                            // ...visible to all waves
        if (wave == 0 && k + KT < KDIM)             // kick DMA for next K slice
            tdm_load_w_tile(W + (size_t)n0 * KDIM + (k + KT), &ldsWf[buf ^ 1][0]);
        // cooperative f32 -> packed bf16 conversion of the landed tile
        {
            const int r  = tid >> 2;                // 0..63 row
            const int ch = (tid & 3) * 8;           // 8-float K sub-chunk
            const float* src = &ldsWf[buf][r * KT + ch];
            float4 f0 = *(const float4*)(src);
            float4 f1 = *(const float4*)(src + 4);
            uint4 pk;
            pk.x = pack2bf(f0.x, f0.y);
            pk.y = pack2bf(f0.z, f0.w);
            pk.z = pack2bf(f1.x, f1.y);
            pk.w = pack2bf(f1.z, f1.w);
            *(uint4*)&ldsWb[r * (KT / 2) + (tid & 3) * 4] = pk;
        }
#else
        __syncthreads();                            // prior reads done
        // manual staging: global load + convert + LDS store
        {
            const int r  = tid >> 2;
            const int ch = (tid & 3) * 8;
            const float* wrow = W + (size_t)(n0 + r) * KDIM + k + ch;
            float4 f0 = *(const float4*)(wrow);
            float4 f1 = *(const float4*)(wrow + 4);
            if (k + KT < KDIM) __builtin_prefetch(wrow + KT, 0, 1);
            uint4 pk;
            pk.x = pack2bf(f0.x, f0.y);
            pk.y = pack2bf(f0.z, f0.w);
            pk.z = pack2bf(f1.x, f1.y);
            pk.w = pack2bf(f1.z, f1.w);
            *(uint4*)&ldsWb[r * (KT / 2) + (tid & 3) * 4] = pk;
        }
#endif

        // ---- A fragment straight from global (convert f32 -> bf16) --------
        BF16x16 afrag;
        {
            const float* ap = arow + k + hi * 8;
            float4 a0 = *(const float4*)(ap);
            float4 a1 = *(const float4*)(ap + 4);
            float4 a2 = *(const float4*)(ap + 16);
            float4 a3 = *(const float4*)(ap + 20);
            if (k + KT < KDIM) __builtin_prefetch(ap + KT, 0, 1);  // global_prefetch_b8
            afrag.u[0] = pack2bf(a0.x, a0.y);
            afrag.u[1] = pack2bf(a0.z, a0.w);
            afrag.u[2] = pack2bf(a1.x, a1.y);
            afrag.u[3] = pack2bf(a1.z, a1.w);
            afrag.u[4] = pack2bf(a2.x, a2.y);
            afrag.u[5] = pack2bf(a2.z, a2.w);
            afrag.u[6] = pack2bf(a3.x, a3.y);
            afrag.u[7] = pack2bf(a3.z, a3.w);
        }

        __syncthreads();  // ldsWb ready

        // ---- 4 WMMA tiles across the 64-wide N strip ----------------------
        // B-matrix 32x16 bf16: lane holds column N = lane%16, K chunks
        // [hi*8, hi*8+8) and [16+hi*8, +8) (same K order as A).
#pragma unroll
        for (int t = 0; t < 4; ++t) {
            BF16x16 bfrag;
            const unsigned int* bp =
                &ldsWb[(t * 16 + lo16) * (KT / 2) + hi * 4];
            uint4 b0 = *(const uint4*)(bp);
            uint4 b1 = *(const uint4*)(bp + 8);
            bfrag.u[0] = b0.x; bfrag.u[1] = b0.y;
            bfrag.u[2] = b0.z; bfrag.u[3] = b0.w;
            bfrag.u[4] = b1.x; bfrag.u[5] = b1.y;
            bfrag.u[6] = b1.z; bfrag.u[7] = b1.w;
            acc[t] = __builtin_amdgcn_wmma_f32_16x16x32_bf16(
                /*neg_a=*/false, afrag.v,
                /*neg_b=*/false, bfrag.v,
                /*c_mod=*/(short)0, acc[t],
                /*reuse_a=*/false, /*reuse_b=*/false);
        }
        // next iteration's leading barrier orders these reads vs. new writes
    }

    // ---- epilogue: add bias, store f32 ------------------------------------
    // C/D layout: VGPR r -> lanes 0-15: (M=r, N=lane) ; lanes 16-31: (M=r+8).
#pragma unroll
    for (int t = 0; t < 4; ++t) {
        const int col = n0 + t * 16 + lo16;
        const float bv = bias[col];
#pragma unroll
        for (int r = 0; r < 8; ++r) {
            const int row = m0 + r + 8 * hi;
            out[(size_t)row * NDIM + col] = acc[t][r] + bv;
        }
    }
}

// ---------------------------------------------------------------------------
// Quantum head, in place over proj: per 8-group j=0..7,
//   c[j] = cos(proj[j] + theta[j])
//   expz[0] = c1*...*c7 ; expz[j>=1] = c0*...*cj
// One thread per head-group (8 contiguous floats), fully coalesced b128 I/O.
// ---------------------------------------------------------------------------
__global__ __launch_bounds__(256)
void quantum_head_kernel(float* __restrict__ p, const float* __restrict__ theta)
{
    const size_t g    = (size_t)blockIdx.x * blockDim.x + threadIdx.x;
    const size_t base = g * 8;

    float4 a0 = *(const float4*)(p + base);
    float4 a1 = *(const float4*)(p + base + 4);

    float c0 = __cosf(a0.x + theta[0]);
    float c1 = __cosf(a0.y + theta[1]);
    float c2 = __cosf(a0.z + theta[2]);
    float c3 = __cosf(a0.w + theta[3]);
    float c4 = __cosf(a1.x + theta[4]);
    float c5 = __cosf(a1.y + theta[5]);
    float c6 = __cosf(a1.z + theta[6]);
    float c7 = __cosf(a1.w + theta[7]);

    // prefix products (wires 1..7)
    float e1 = c0 * c1;
    float e2 = e1 * c2;
    float e3 = e2 * c3;
    float e4 = e3 * c4;
    float e5 = e4 * c5;
    float e6 = e5 * c6;
    float e7 = e6 * c7;
    // wire 0: product of c1..c7 (no division to stay safe at c0 == 0)
    float s  = c1 * c2;
    s *= c3; s *= c4; s *= c5; s *= c6; s *= c7;

    float4 o0 = make_float4(s,  e1, e2, e3);
    float4 o1 = make_float4(e4, e5, e6, e7);
    *(float4*)(p + base)     = o0;
    *(float4*)(p + base + 4) = o1;
}

// ---------------------------------------------------------------------------
extern "C" void kernel_launch(void* const* d_in, const int* in_sizes, int n_in,
                              void* d_out, int out_size, void* d_ws, size_t ws_size,
                              hipStream_t stream)
{
    const float* x      = (const float*)d_in[0];  // [8,512,768]
    const float* W_proj = (const float*)d_in[1];  // [768,768]
    const float* b_proj = (const float*)d_in[2];  // [768]
    const float* theta  = (const float*)d_in[3];  // [8]
    const float* W_comb = (const float*)d_in[4];  // [768,768]
    const float* b_comb = (const float*)d_in[5];  // [768]
    float*       out    = (float*)d_out;          // [8,512,768]
    float*       proj   = (float*)d_ws;           // 4096*768 f32 = 12.6 MB scratch

    dim3 block(256);
    dim3 grid(NDIM / NT, MDIM / MT);              // (12, 32)

    // 1) proj = x @ Wp^T + bp
    gemm_bias_wmma<<<grid, block, 0, stream>>>(x, W_proj, b_proj, proj);

    // 2) quantum head in place: proj -> expz
    const int groups = MDIM * (NDIM / 8);         // 393216
    quantum_head_kernel<<<dim3(groups / 256), block, 0, stream>>>(proj, theta);

    // 3) out = expz @ Wc^T + bc
    gemm_bias_wmma<<<grid, block, 0, stream>>>(proj, W_comb, b_comb, out);
}
